// GradientGCN_85091892068782
// MI455X (gfx1250) — compile-verified, matrix operand order checked by reading, not run on hardware
//
#include <hip/hip_runtime.h>
#include <math.h>

// ---------------- problem constants (from reference) ----------------
#define Nn 50000
#define Ee 500000
#define Hd 128
#define NHd 8
#define HCd 64
#define NLd 2
#define NCd 2

// ---------------- WMMA types ----------------
typedef __attribute__((ext_vector_type(16))) __bf16 v16bf;
typedef __attribute__((ext_vector_type(8)))  float  v8f;

__device__ __forceinline__ unsigned short f2bf(float f) {
    union { float f; unsigned u; } x; x.f = f;
    unsigned u = x.u;
    unsigned r = u + 0x7FFFu + ((u >> 16) & 1u);   // RNE
    return (unsigned short)(r >> 16);
}

__device__ __forceinline__ float softplusf(float x) {
    return x > 20.f ? x : log1pf(expf(x));
}

// monotone float<->uint encoding for atomicMax-based segment max
__device__ __forceinline__ unsigned enc_f32(float f) {
    unsigned u = __float_as_uint(f);
    return (u & 0x80000000u) ? ~u : (u | 0x80000000u);
}
__device__ __forceinline__ float dec_f32(unsigned e) {
    unsigned u = (e & 0x80000000u) ? (e ^ 0x80000000u) : ~e;
    return __uint_as_float(u);
}
#define M_NEG_INF_ENC 0x007FFFFFu   // enc(-inf)

// Load one 16x32 bf16 WMMA operand fragment from LDS.
// Row-major rows of length `stride` (stride multiple of 8 elems -> 16B aligned).
// Per ISA 7.12.2: lanes 0-15 take K {k0..k0+7, k0+16..k0+23},
//                 lanes 16-31 take K {k0+8..k0+15, k0+24..k0+31}.
__device__ __forceinline__ v16bf ldfrag(const unsigned short* base, int row,
                                        int stride, int k0, int laneHi) {
    const unsigned short* p = base + row * stride + k0 + (laneHi ? 8 : 0);
    union { uint4 u[2]; v16bf v; } c;
    c.u[0] = *(const uint4*)(p);
    c.u[1] = *(const uint4*)(p + 16);
    return c.v;
}

#define WMMA_BF16(a, b, c) \
    __builtin_amdgcn_wmma_f32_16x16x32_bf16(false, (a), false, (b), (short)0, (c), false, false)

// ---------------- utility kernels ----------------
__global__ void k_fill_u32(unsigned* __restrict__ p, unsigned v, long long n) {
    long long i = (long long)blockIdx.x * blockDim.x + threadIdx.x;
    if (i < n) p[i] = v;
}
__global__ void k_copy_f32(float* __restrict__ d, const float* __restrict__ s, long long n) {
    long long i = (long long)blockIdx.x * blockDim.x + threadIdx.x;
    if (i < n) d[i] = s[i];
}

// ---------------- coord basis (cd, cc, cv per edge) ----------------
__global__ void k_basis(const float* __restrict__ pp, const int* __restrict__ src,
                        const int* __restrict__ dst, float* __restrict__ basis) {
    int e = blockIdx.x * blockDim.x + threadIdx.x;
    if (e >= Ee) return;
    int si = src[e], di = dst[e];
    float sx = pp[si*3+0], sy = pp[si*3+1], sz = pp[si*3+2];
    float dx = pp[di*3+0], dy = pp[di*3+1], dz = pp[di*3+2];
    float cdx = sx - dx, cdy = sy - dy, cdz = sz - dz;
    float radial = cdx*cdx + cdy*cdy + cdz*cdz;
    float ccx = sy*dz - sz*dy, ccy = sz*dx - sx*dz, ccz = sx*dy - sy*dx;
    float i1 = 1.f / (sqrtf(radial) + 1.f);
    cdx *= i1; cdy *= i1; cdz *= i1;
    float i2 = 1.f / (sqrtf(ccx*ccx + ccy*ccy + ccz*ccz) + 1.f);
    ccx *= i2; ccy *= i2; ccz *= i2;
    float cvx = cdy*ccz - cdz*ccy, cvy = cdz*ccx - cdx*ccz, cvz = cdx*ccy - cdy*ccx;
    float* b = basis + (size_t)e * 9;
    b[0]=cdx; b[1]=cdy; b[2]=cdz; b[3]=ccx; b[4]=ccy; b[5]=ccz; b[6]=cvx; b[7]=cvy; b[8]=cvz;
}

// ---------------- Y[M,128] = X[M,128] @ W[128,128]  (bf16 WMMA, f32 acc) ----------------
__global__ __launch_bounds__(128) void k_gemm128(const float* __restrict__ X,
                                                 const float* __restrict__ W,
                                                 float* __restrict__ Y, int M) {
    __shared__ __align__(16) unsigned short sWt[128 * 136];     // W transposed, bf16
    __shared__ __align__(16) unsigned short sXt[4][16 * 136];   // per-wave 16-row A tile
    int t = threadIdx.x;
    int row0 = blockIdx.x * 64;
    for (int i = t; i < 128 * 128; i += 128) {
        int k = i >> 7, n = i & 127;
        sWt[n * 136 + k] = f2bf(W[k * 128 + n]);                // coalesced global read
    }
    for (int i = t; i < 64 * 128; i += 128) {
        int r = i >> 7, c = i & 127;
        int row = row0 + r; if (row >= M) row = M - 1;
        sXt[r >> 4][(r & 15) * 136 + c] = f2bf(X[(size_t)row * 128 + c]);
    }
    __syncthreads();
    int lane = t & 31, w = t >> 5, laneHi = lane >> 4, nlo = lane & 15;
    const unsigned short* A = sXt[w];
    v8f z = {0,0,0,0,0,0,0,0};
    v8f acc[8];
    for (int nt = 0; nt < 8; ++nt) acc[nt] = z;
    for (int k0 = 0; k0 < 128; k0 += 32) {
        v16bf a = ldfrag(A, nlo, 136, k0, laneHi);
        for (int nt = 0; nt < 8; ++nt) {
            v16bf b = ldfrag(sWt, nt * 16 + nlo, 136, k0, laneHi);
            acc[nt] = WMMA_BF16(a, b, acc[nt]);
        }
    }
    for (int nt = 0; nt < 8; ++nt)
        for (int r = 0; r < 8; ++r) {
            int mrow = row0 + w * 16 + r + laneHi * 8;
            if (mrow < M) Y[(size_t)mrow * 128 + nt * 16 + nlo] = acc[nt][r];
        }
}

// ---------------- edge scores: keff = edge_attr@We (WMMA) + K[src]; q·k; segment max ----------------
__global__ __launch_bounds__(64) void k_edge_score(const float* __restrict__ edge_attr,
                                                   const float* __restrict__ Wep,
                                                   const float* __restrict__ Qm,
                                                   const float* __restrict__ Km,
                                                   const int* __restrict__ src,
                                                   const int* __restrict__ dst,
                                                   float* __restrict__ scores,
                                                   unsigned* __restrict__ m_enc) {
    __shared__ __align__(16) unsigned short sWt[128 * 136];     // We^T bf16 (34816 B)
    __shared__ __align__(16) unsigned char  sBuf[2][16 * 264 * 2]; // 8448 B per wave (A tile, then keff)
    int t = threadIdx.x;
    int e0 = blockIdx.x * 32;
    for (int i = t; i < 128 * 128; i += 64) {
        int k = i >> 7, n = i & 127;
        sWt[n * 136 + k] = f2bf(Wep[k * 128 + n]);
    }
    for (int i = t; i < 32 * 128; i += 64) {
        int r = i >> 7, c = i & 127;
        int e = e0 + r; if (e >= Ee) e = Ee - 1;
        ((unsigned short*)sBuf[r >> 4])[(r & 15) * 136 + c] = f2bf(edge_attr[(size_t)e * 128 + c]);
    }
    __syncthreads();
    int lane = t & 31, w = t >> 5, laneHi = lane >> 4, nlo = lane & 15;
    const unsigned short* A = (const unsigned short*)sBuf[w];
    v8f z = {0,0,0,0,0,0,0,0};
    v8f acc[8];
    for (int nt = 0; nt < 8; ++nt) acc[nt] = z;
    for (int k0 = 0; k0 < 128; k0 += 32) {
        v16bf a = ldfrag(A, nlo, 136, k0, laneHi);
        for (int nt = 0; nt < 8; ++nt) {
            v16bf b = ldfrag(sWt, nt * 16 + nlo, 136, k0, laneHi);
            acc[nt] = WMMA_BF16(a, b, acc[nt]);
        }
    }
    // A tile fully consumed -> park keff (edge_attr@We) over it, f32 stride 132
    float* keff = (float*)sBuf[w];
    for (int nt = 0; nt < 8; ++nt)
        for (int r = 0; r < 8; ++r)
            keff[(r + laneHi * 8) * 132 + nt * 16 + nlo] = acc[nt][r];
    __syncthreads();
    // 16 edges x 8 heads per wave -> 4 items per lane
    for (int ii = 0; ii < 4; ++ii) {
        int it = lane + 32 * ii;
        int el = it >> 3, h = it & 7;
        int e = e0 + w * 16 + el;
        if (e < Ee) {
            int si = src[e], di = dst[e];
            const float* qrow = Qm + (size_t)di * 128 + h * 16;
            const float* krow = Km + (size_t)si * 128 + h * 16;
            const float* kf   = keff + el * 132 + h * 16;
            float sc = 0.f;
            for (int j = 0; j < 16; ++j) sc += qrow[j] * (krow[j] + kf[j]);
            sc *= 0.25f;                                  // 1/sqrt(16)
            scores[(size_t)e * 8 + h] = sc;
            atomicMax(m_enc + (size_t)di * 8 + h, enc_f32(sc));
        }
    }
}

// ---------------- e = exp(score - m[dst]); s[dst] += e ----------------
__global__ void k_soft_e(float* __restrict__ scores, const unsigned* __restrict__ m_enc,
                         float* __restrict__ ssum, const int* __restrict__ dst) {
    int i = blockIdx.x * blockDim.x + threadIdx.x;
    if (i >= Ee * NHd) return;
    int e = i >> 3, h = i & 7;
    int di = dst[e];
    float mu = dec_f32(m_enc[(size_t)di * 8 + h]);
    if (!(mu > -1e38f && mu < 1e38f)) mu = 0.f;           // isfinite guard (reference)
    float ev = expf(scores[i] - mu);
    scores[i] = ev;
    atomicAdd(ssum + (size_t)di * 8 + h, ev);
}

// ---------------- agg[dst] += alpha * V[src] ----------------
__global__ void k_agg(const float* __restrict__ scores, const float* __restrict__ ssum,
                      const float* __restrict__ V, const int* __restrict__ src,
                      const int* __restrict__ dst, float* __restrict__ agg) {
    long long tid = (long long)blockIdx.x * blockDim.x + threadIdx.x;
    if (tid >= (long long)Ee * 128) return;
    int e = (int)(tid >> 7), j = (int)(tid & 127), h = j >> 4;
    int si = src[e], di = dst[e];
    float alpha = scores[(size_t)e * 8 + h] / (ssum[(size_t)di * 8 + h] + 1e-16f);
    atomicAdd(agg + (size_t)di * 128 + j, alpha * V[(size_t)si * 128 + j]);
}

// ---------------- Y = softplus?( LN( x + agg@Wo ) * g + b ) ----------------
__global__ __launch_bounds__(128) void k_node_out(const float* __restrict__ agg,
                                                  const float* __restrict__ Wo,
                                                  const float* __restrict__ xres,
                                                  const float* __restrict__ g,
                                                  const float* __restrict__ b,
                                                  float* __restrict__ Y, int M, int do_sp) {
    __shared__ __align__(16) unsigned short sWt[128 * 136];
    __shared__ __align__(16) unsigned short sXt[4][16 * 136];
    int t = threadIdx.x;
    int row0 = blockIdx.x * 64;
    for (int i = t; i < 128 * 128; i += 128) {
        int k = i >> 7, n = i & 127;
        sWt[n * 136 + k] = f2bf(Wo[k * 128 + n]);
    }
    for (int i = t; i < 64 * 128; i += 128) {
        int r = i >> 7, c = i & 127;
        int row = row0 + r; if (row >= M) row = M - 1;
        sXt[r >> 4][(r & 15) * 136 + c] = f2bf(agg[(size_t)row * 128 + c]);
    }
    __syncthreads();
    int lane = t & 31, w = t >> 5, laneHi = lane >> 4, nlo = lane & 15;
    const unsigned short* A = sXt[w];
    v8f z = {0,0,0,0,0,0,0,0};
    v8f acc[8];
    for (int nt = 0; nt < 8; ++nt) acc[nt] = z;
    for (int k0 = 0; k0 < 128; k0 += 32) {
        v16bf a = ldfrag(A, nlo, 136, k0, laneHi);
        for (int nt = 0; nt < 8; ++nt) {
            v16bf bf = ldfrag(sWt, nt * 16 + nlo, 136, k0, laneHi);
            acc[nt] = WMMA_BF16(a, bf, acc[nt]);
        }
    }
    // residual add
    for (int nt = 0; nt < 8; ++nt)
        for (int r = 0; r < 8; ++r) {
            int mrow = row0 + w * 16 + r + laneHi * 8;
            int rr = (mrow < M) ? mrow : (M - 1);
            acc[nt][r] += xres[(size_t)rr * 128 + nt * 16 + nlo];
        }
    // layernorm per row; C-layout row split matches shfl_xor masks 1..8 (halves independent)
    for (int r = 0; r < 8; ++r) {
        float s1 = 0.f;
        for (int nt = 0; nt < 8; ++nt) s1 += acc[nt][r];
        for (int msk = 1; msk <= 8; msk <<= 1) s1 += __shfl_xor(s1, msk, 32);
        float mu = s1 * (1.0f / 128.0f);
        float s2 = 0.f;
        for (int nt = 0; nt < 8; ++nt) { float d = acc[nt][r] - mu; s2 += d * d; }
        for (int msk = 1; msk <= 8; msk <<= 1) s2 += __shfl_xor(s2, msk, 32);
        float rs = rsqrtf(s2 * (1.0f / 128.0f) + 1e-5f);
        int mrow = row0 + w * 16 + r + laneHi * 8;
        if (mrow < M) {
            for (int nt = 0; nt < 8; ++nt) {
                int n = nt * 16 + nlo;
                float o = (acc[nt][r] - mu) * rs * g[n] + b[n];
                if (do_sp) o = softplusf(o);
                Y[(size_t)mrow * 128 + n] = o;
            }
        }
    }
}

// ---------------- gradient: coff = sp(ef@W1+b1)@W2+b2; grad[dst] += coff·basis ----------------
__global__ __launch_bounds__(64) void k_grad(const float* __restrict__ nf,
                                             const float* __restrict__ edge_attr,
                                             const int* __restrict__ src,
                                             const int* __restrict__ dst,
                                             const float* __restrict__ basis,
                                             const float* __restrict__ W1,
                                             const float* __restrict__ b1,
                                             const float* __restrict__ W2,
                                             const float* __restrict__ b2,
                                             float* __restrict__ grad) {
    __shared__ __align__(16) unsigned short sW1t[64 * 264];        // W1^T bf16 (33792 B)
    __shared__ __align__(16) unsigned char  sBuf[2][16 * 264 * 2]; // A tile, then h1 (8448 B each)
    int t = threadIdx.x;
    int e0 = blockIdx.x * 32;
    for (int i = t; i < 256 * 64; i += 64) {
        int k = i >> 6, n = i & 63;
        sW1t[n * 264 + k] = f2bf(W1[k * 64 + n]);
    }
    for (int i = t; i < 32 * 256; i += 64) {
        int r = i >> 8, c = i & 255;
        int e = e0 + r; if (e >= Ee) e = Ee - 1;
        float v;
        if (c < 128) {
            int si = src[e], di = dst[e];
            v = nf[(size_t)si * 128 + c] * nf[(size_t)di * 128 + c];
        } else {
            v = edge_attr[(size_t)e * 128 + (c - 128)];
        }
        ((unsigned short*)sBuf[r >> 4])[(r & 15) * 264 + c] = f2bf(v);
    }
    __syncthreads();
    int lane = t & 31, w = t >> 5, laneHi = lane >> 4, nlo = lane & 15;
    const unsigned short* A = (const unsigned short*)sBuf[w];
    v8f z = {0,0,0,0,0,0,0,0};
    v8f acc[4];
    for (int nt = 0; nt < 4; ++nt) acc[nt] = z;
    for (int k0 = 0; k0 < 256; k0 += 32) {
        v16bf a = ldfrag(A, nlo, 264, k0, laneHi);
        for (int nt = 0; nt < 4; ++nt) {
            v16bf bf = ldfrag(sW1t, nt * 16 + nlo, 264, k0, laneHi);
            acc[nt] = WMMA_BF16(a, bf, acc[nt]);
        }
    }
    // bias + softplus -> h1 in LDS (A consumed), stride 66 f32
    float* h1 = (float*)sBuf[w];
    for (int nt = 0; nt < 4; ++nt) {
        float bb = b1[nt * 16 + nlo];
        for (int r = 0; r < 8; ++r)
            h1[(r + laneHi * 8) * 66 + nt * 16 + nlo] = softplusf(acc[nt][r] + bb);
    }
    __syncthreads();
    if (lane < 16) {
        int e = e0 + w * 16 + lane;
        if (e < Ee) {
            float c0 = b2[0], c1 = b2[1], c2 = b2[2];
            const float* hr = h1 + lane * 66;
            for (int j = 0; j < 64; ++j) {
                float hv = hr[j];
                c0 += hv * W2[j * 3 + 0];
                c1 += hv * W2[j * 3 + 1];
                c2 += hv * W2[j * 3 + 2];
            }
            const float* bs = basis + (size_t)e * 9;
            float gx = c0 * bs[0] + c1 * bs[3] + c2 * bs[6];
            float gy = c0 * bs[1] + c1 * bs[4] + c2 * bs[7];
            float gz = c0 * bs[2] + c1 * bs[5] + c2 * bs[8];
            int di = dst[e];
            atomicAdd(grad + (size_t)di * 3 + 0, gx);
            atomicAdd(grad + (size_t)di * 3 + 1, gy);
            atomicAdd(grad + (size_t)di * 3 + 2, gz);
        }
    }
}

// ---------------- host orchestration ----------------
extern "C" void kernel_launch(void* const* d_in, const int* in_sizes, int n_in,
                              void* d_out, int out_size, void* d_ws, size_t ws_size,
                              hipStream_t stream) {
    const float* node_attr = (const float*)d_in[0];
    const float* edge_attr = (const float*)d_in[1];
    const float* pert_pos  = (const float*)d_in[2];
    const float* Wq = (const float*)d_in[3];
    const float* Wk = (const float*)d_in[4];
    const float* Wv = (const float*)d_in[5];
    const float* We = (const float*)d_in[6];
    const float* Wo = (const float*)d_in[7];
    const float* ln_g = (const float*)d_in[8];
    const float* ln_b = (const float*)d_in[9];
    const float* dm_W1 = (const float*)d_in[10];
    const float* dm_b1 = (const float*)d_in[11];
    const float* dm_W2 = (const float*)d_in[12];
    const float* dm_b2 = (const float*)d_in[13];
    const int*   eidx  = (const int*)d_in[14];
    const int* src = eidx;
    const int* dst = eidx + Ee;

    float* out_nf   = (float*)d_out;
    float* out_grad = out_nf + (size_t)Nn * Hd;

    // workspace carve
    size_t off = 0;
    char* wsb = (char*)d_ws;
    auto carve = [&](size_t bytes) -> char* {
        char* p = wsb + off;
        off += (bytes + 255) & ~(size_t)255;
        return p;
    };
    float*    xbuf   = (float*)carve((size_t)Nn * Hd * 4);
    float*    xalt   = (float*)carve((size_t)Nn * Hd * 4);
    float*    Qm     = (float*)carve((size_t)Nn * Hd * 4);
    float*    Km     = (float*)carve((size_t)Nn * Hd * 4);
    float*    Vm     = (float*)carve((size_t)Nn * Hd * 4);
    float*    aggm   = (float*)carve((size_t)Nn * Hd * 4);
    unsigned* m_enc  = (unsigned*)carve((size_t)Nn * NHd * 4);
    float*    ssum   = (float*)carve((size_t)Nn * NHd * 4);
    float*    scores = (float*)carve((size_t)Ee * NHd * 4);
    float*    basis  = (float*)carve((size_t)Ee * 9 * 4);

    const int nodeGemmBlocks = (Nn + 63) / 64;
    const int edgeTileBlocks = (Ee + 31) / 32;

    // basis vectors (shared by both outer layers)
    k_basis<<<(Ee + 255) / 256, 256, 0, stream>>>(pert_pos, src, dst, basis);
    // x <- node_attr ; gradient <- 0
    k_copy_f32<<<((long long)Nn * Hd + 255) / 256, 256, 0, stream>>>(xbuf, node_attr, (long long)Nn * Hd);
    k_fill_u32<<<((long long)Nn * 3 + 255) / 256, 256, 0, stream>>>((unsigned*)out_grad, 0u, (long long)Nn * 3);

    for (int mi = 0; mi < NLd; ++mi) {
        for (int ci = 0; ci < NCd; ++ci) {
            size_t woff = (size_t)(mi * NCd + ci) * Hd * Hd;
            size_t loff = (size_t)(mi * NCd + ci) * Hd;
            // projections
            k_gemm128<<<nodeGemmBlocks, 128, 0, stream>>>(xbuf, Wq + woff, Qm, Nn);
            k_gemm128<<<nodeGemmBlocks, 128, 0, stream>>>(xbuf, Wk + woff, Km, Nn);
            k_gemm128<<<nodeGemmBlocks, 128, 0, stream>>>(xbuf, Wv + woff, Vm, Nn);
            // reset segment buffers
            k_fill_u32<<<((long long)Nn * NHd + 255) / 256, 256, 0, stream>>>(m_enc, M_NEG_INF_ENC, (long long)Nn * NHd);
            k_fill_u32<<<((long long)Nn * NHd + 255) / 256, 256, 0, stream>>>((unsigned*)ssum, 0u, (long long)Nn * NHd);
            k_fill_u32<<<((long long)Nn * Hd + 255) / 256, 256, 0, stream>>>((unsigned*)aggm, 0u, (long long)Nn * Hd);
            // attention
            k_edge_score<<<edgeTileBlocks, 64, 0, stream>>>(edge_attr, We + woff, Qm, Km, src, dst, scores, m_enc);
            k_soft_e<<<((long long)Ee * NHd + 255) / 256, 256, 0, stream>>>(scores, m_enc, ssum, dst);
            k_agg<<<(int)(((long long)Ee * Hd + 255) / 256), 256, 0, stream>>>(scores, ssum, Vm, src, dst, aggm);
            // output projection + residual + LN (+softplus between convs)
            k_node_out<<<nodeGemmBlocks, 128, 0, stream>>>(aggm, Wo + woff, xbuf,
                                                           ln_g + loff, ln_b + loff,
                                                           xalt, Nn, (ci < NCd - 1) ? 1 : 0);
            float* tmp = xbuf; xbuf = xalt; xalt = tmp;
        }
        // per-layer gradient contribution (nf == current xbuf)
        k_grad<<<edgeTileBlocks, 64, 0, stream>>>(xbuf, edge_attr, src, dst, basis,
                                                  dm_W1 + (size_t)mi * 2 * Hd * HCd,
                                                  dm_b1 + (size_t)mi * HCd,
                                                  dm_W2 + (size_t)mi * HCd * 3,
                                                  dm_b2 + (size_t)mi * 3,
                                                  out_grad);
    }
    // nf output
    k_copy_f32<<<((long long)Nn * Hd + 255) / 256, 256, 0, stream>>>(out_nf, xbuf, (long long)Nn * Hd);
}
